// MyVGG_83253646066026
// MI455X (gfx1250) — matrix operand order repaired and checked by reading, not compiled, not run on hardware
//
#include <hip/hip_runtime.h>

// ---------------------------------------------------------------------------
// MI455X reasoning:
//  * x stream (335 MB) is the roofline: ~14.4us @ 23.3 TB/s. Kernel `gene_kernel`
//    reads each 128B line of x exactly once (dense per-row feature windows),
//    fixed-pattern segment widths -> closed-form offsets, float4 loads.
//  * All GEMMs (~1.3 GFLOP) are fused into one wave32 WMMA kernel using
//    v_wmma_f32_16x16x32_f16: 16-row tile per wave, weights pre-swizzled into
//    a contiguous-k B-matrix lane layout so every operand load is a 16B/32B b128.
// ---------------------------------------------------------------------------

#define B_ROWS 4096
#define F_FEAT 20480
#define G_GENE 1024
#define ML_LEN 512

typedef __attribute__((ext_vector_type(16))) _Float16 v16h;
typedef __attribute__((ext_vector_type(8)))  _Float16 v8h;
typedef __attribute__((ext_vector_type(8)))  float    v8f;

// --- f32 -> f16 elementwise conversion (z1/z2) ------------------------------
__global__ void cvt_f16_kernel(const float* __restrict__ in,
                               _Float16* __restrict__ out, int n) {
    for (int i = blockIdx.x * blockDim.x + threadIdx.x; i < n;
         i += gridDim.x * blockDim.x)
        out[i] = (_Float16)in[i];
}

// --- Pre-swizzle a [K,64] f32 weight matrix into WMMA B-matrix lane layout --
// B 16-bit 32x16 layout, contiguous-k order (same as A frags): lane L holds col = (L&15) of ntile,
// K = kt*32 + 16*(L>>4) + j, j=0..15 -> 16 contiguous f16 (32B) per lane.
// grid = K/32 blocks, 128 threads (kt = blockIdx, nt = tid>>5, L = tid&31).
__global__ void swizzleB_kernel(const float* __restrict__ W,
                                _Float16* __restrict__ out) {
    int kt = blockIdx.x;
    int nt = (threadIdx.x >> 5) & 3;
    int L  = threadIdx.x & 31;
    int col   = nt * 16 + (L & 15);
    int kbase = kt * 32 + ((L >> 4) << 4);
    int obase = (((kt << 2) + nt) * 32 + L) * 16;
#pragma unroll
    for (int j = 0; j < 16; ++j)
        out[obase + j] = (_Float16)W[(kbase + j) * 64 + col];
}

// --- gene = relu(block-diag(x * w_gene) + b_gene), stored f16 [B, 1024] -----
// Widths tile [8,16,24,32]: off(g) = 80*(g/4) + 4*r*(r+1), w = 8*(r+1), r=g&3.
// Wave = 1 row x 32 genes (640 contiguous floats of that row): every HBM line
// is fully consumed inside the wave. 256 threads = 8 rows.
__global__ void __launch_bounds__(256)
gene_kernel(const float* __restrict__ x, const float* __restrict__ wg,
            const float* __restrict__ bg, _Float16* __restrict__ geneh) {
    int lane = threadIdx.x & 31;
    int wv   = threadIdx.x >> 5;
    int b = blockIdx.x * 8 + wv;
    int g = blockIdx.y * 32 + lane;
    int r = g & 3;
    int off = (g >> 2) * 80 + 4 * r * (r + 1);   // feature start (mult of 8)
    int w   = 8 * (r + 1);                       // width (mult of 4)
    const float* xr = x + (size_t)b * F_FEAT + off;
    const float* wr = wg + off;
    float s = 0.f;
    for (int i = 0; i < w; i += 4) {
        float4 xv = *(const float4*)(xr + i);
        float4 wv4 = *(const float4*)(wr + i);
        s += xv.x * wv4.x + xv.y * wv4.y + xv.z * wv4.z + xv.w * wv4.w;
    }
    s += bg[g];
    s = s > 0.f ? s : 0.f;
    geneh[(size_t)b * G_GENE + g] = (_Float16)s;
}

// --- WMMA helpers -----------------------------------------------------------
__device__ inline v16h make_frag(const _Float16* p) {
    // two contiguous 16B chunks: elements p[0..7], p[8..15]
    v8h lo = *(const v8h*)p;
    v8h hi = *(const v8h*)(p + 8);
    v16h rr;
#pragma unroll
    for (int i = 0; i < 8; ++i) { rr[i] = lo[i]; rr[8 + i] = hi[i]; }
    return rr;
}

#define WMMA16(a, b, c) \
    __builtin_amdgcn_wmma_f32_16x16x32_f16(false, (a), false, (b), (short)0, (c), false, false)

// A tile (16 rows, stride==K, f16) x pre-swizzled B [K,64] -> relu -> LDS f16
__device__ inline void gemm_bias_relu_to_lds(
    const _Float16* __restrict__ A, int K, const _Float16* __restrict__ Bsw,
    const float* __restrict__ bias, _Float16* ldsDst, int dstStride,
    int colBase, int lane) {
    v8f acc[4];
#pragma unroll
    for (int nt = 0; nt < 4; ++nt)
#pragma unroll
        for (int i = 0; i < 8; ++i) acc[nt][i] = 0.f;

    int m = lane & 15;             // A row / D col-in-ntile
    int s = (lane >> 4) << 4;      // A K-offset select (0 or 16)
    for (int kt = 0; kt < (K >> 5); ++kt) {
        v16h a = make_frag(A + m * K + (kt << 5) + s);
#pragma unroll
        for (int nt = 0; nt < 4; ++nt) {
            v16h b = make_frag(Bsw + ((((kt << 2) + nt) << 5) + lane) * 16);
            acc[nt] = WMMA16(a, b, acc[nt]);
        }
    }
    int hiRow = (lane >> 4) << 3;
#pragma unroll
    for (int nt = 0; nt < 4; ++nt) {
        float bv = bias[nt * 16 + m];
#pragma unroll
        for (int r = 0; r < 8; ++r) {
            float v = acc[nt][r] + bv;
            v = v > 0.f ? v : 0.f;
            ldsDst[(r + hiRow) * dstStride + colBase + nt * 16 + m] = (_Float16)v;
        }
    }
}

// --- Fused MLP: [h2|h1|h0] -> @Wa relu -> @Wb relu -> @Wc + bc --------------
// 128 threads = 4 waves; each wave owns a 16-row batch tile end-to-end.
__global__ void __launch_bounds__(128)
mlp_kernel(const _Float16* __restrict__ geneh, const _Float16* __restrict__ z1h,
           const _Float16* __restrict__ z2h,
           const _Float16* __restrict__ W0s, const _Float16* __restrict__ W1s,
           const _Float16* __restrict__ W2s, const _Float16* __restrict__ Was,
           const _Float16* __restrict__ Wbs,
           const float* __restrict__ b0, const float* __restrict__ b1,
           const float* __restrict__ b2, const float* __restrict__ ba,
           const float* __restrict__ bb, const float* __restrict__ Wc,
           const float* __restrict__ bc, float* __restrict__ out) {
    __shared__ _Float16 lds_d2[4 * 16 * 192];   // per-wave d2 = [h2|h1|h0]
    __shared__ _Float16 lds_h[4 * 16 * 64];     // per-wave hidden after Wa

    int lane = threadIdx.x & 31;
    int wv   = threadIdx.x >> 5;
    int tile0 = (blockIdx.x * 4 + wv) * 16;
    _Float16* d2 = lds_d2 + wv * 16 * 192;
    _Float16* hb = lds_h + wv * 16 * 64;

    // d2 columns: [0:64)=h2, [64:128)=h1, [128:192)=h0  (matches concat order)
    gemm_bias_relu_to_lds(z2h  + (size_t)tile0 * ML_LEN, ML_LEN, W2s, b2, d2, 192, 0,   lane);
    gemm_bias_relu_to_lds(z1h  + (size_t)tile0 * ML_LEN, ML_LEN, W1s, b1, d2, 192, 64,  lane);
    gemm_bias_relu_to_lds(geneh + (size_t)tile0 * G_GENE, G_GENE, W0s, b0, d2, 192, 128, lane);
    __syncthreads();

    int m  = lane & 15;
    int s  = (lane >> 4) << 4;
    int hiRow = (lane >> 4) << 3;

    // d2 [16x192] @ Wa [192x64] -> relu -> hb (LDS, A-layout-ready)
    {
        v8f acc[4];
#pragma unroll
        for (int nt = 0; nt < 4; ++nt)
#pragma unroll
            for (int i = 0; i < 8; ++i) acc[nt][i] = 0.f;
#pragma unroll
        for (int kt = 0; kt < 6; ++kt) {
            v16h a = make_frag(d2 + m * 192 + (kt << 5) + s);
#pragma unroll
            for (int nt = 0; nt < 4; ++nt) {
                v16h b = make_frag(Was + ((((kt << 2) + nt) << 5) + lane) * 16);
                acc[nt] = WMMA16(a, b, acc[nt]);
            }
        }
#pragma unroll
        for (int nt = 0; nt < 4; ++nt) {
            float bv = ba[nt * 16 + m];
#pragma unroll
            for (int r = 0; r < 8; ++r) {
                float v = acc[nt][r] + bv;
                v = v > 0.f ? v : 0.f;
                hb[(r + hiRow) * 64 + nt * 16 + m] = (_Float16)v;
            }
        }
    }
    __syncthreads();

    // hb [16x64] @ Wb [64x64] -> relu -> dot(Wc) -> + bc
    {
        v8f acc[4];
#pragma unroll
        for (int nt = 0; nt < 4; ++nt)
#pragma unroll
            for (int i = 0; i < 8; ++i) acc[nt][i] = 0.f;
#pragma unroll
        for (int kt = 0; kt < 2; ++kt) {
            v16h a = make_frag(hb + m * 64 + (kt << 5) + s);
#pragma unroll
            for (int nt = 0; nt < 4; ++nt) {
                v16h b = make_frag(Wbs + ((((kt << 2) + nt) << 5) + lane) * 16);
                acc[nt] = WMMA16(a, b, acc[nt]);
            }
        }
        float part[8];
#pragma unroll
        for (int r = 0; r < 8; ++r) part[r] = 0.f;
#pragma unroll
        for (int nt = 0; nt < 4; ++nt) {
            float bv = bb[nt * 16 + m];
            float wc = Wc[nt * 16 + m];
#pragma unroll
            for (int r = 0; r < 8; ++r) {
                float v = acc[nt][r] + bv;
                v = v > 0.f ? v : 0.f;
                part[r] += v * wc;
            }
        }
        // reduce over the 16 N-lanes (xor masks stay inside each 16-lane group)
#pragma unroll
        for (int mask = 8; mask >= 1; mask >>= 1)
#pragma unroll
            for (int r = 0; r < 8; ++r)
                part[r] += __shfl_xor(part[r], mask, 32);
        if (m == 0) {
            float bcv = bc[0];
#pragma unroll
            for (int r = 0; r < 8; ++r)
                out[tile0 + r + hiRow] = part[r] + bcv;
        }
    }
}

// ---------------------------------------------------------------------------
extern "C" void kernel_launch(void* const* d_in, const int* in_sizes, int n_in,
                              void* d_out, int out_size, void* d_ws, size_t ws_size,
                              hipStream_t stream) {
    const float* x      = (const float*)d_in[0];
    const float* z1     = (const float*)d_in[1];
    const float* z2     = (const float*)d_in[2];
    /* d_in[3] = z3 unused in this branch */
    const float* w_gene = (const float*)d_in[4];
    const float* b_gene = (const float*)d_in[5];
    const float* W0     = (const float*)d_in[6];
    const float* b0     = (const float*)d_in[7];
    const float* W1     = (const float*)d_in[8];
    const float* b1     = (const float*)d_in[9];
    const float* W2     = (const float*)d_in[10];
    const float* b2     = (const float*)d_in[11];
    const float* Wa     = (const float*)d_in[12];
    const float* ba     = (const float*)d_in[13];
    const float* Wb     = (const float*)d_in[14];
    const float* bb     = (const float*)d_in[15];
    const float* Wc     = (const float*)d_in[16];
    const float* bc     = (const float*)d_in[17];
    float* out = (float*)d_out;

    // Workspace layout (all 16B-aligned offsets), ~16.3 MB total
    char* ws = (char*)d_ws;
    _Float16* geneh = (_Float16*)(ws + 0);          // 4096*1024 f16 = 8 MB
    _Float16* z1h   = (_Float16*)(ws + 8388608);    // 4 MB
    _Float16* z2h   = (_Float16*)(ws + 12582912);   // 4 MB
    _Float16* W0s   = (_Float16*)(ws + 16777216);   // 128 KB
    _Float16* W1s   = (_Float16*)(ws + 16908288);   // 64 KB
    _Float16* W2s   = (_Float16*)(ws + 16973824);   // 64 KB
    _Float16* Was   = (_Float16*)(ws + 17039360);   // 24 KB
    _Float16* Wbs   = (_Float16*)(ws + 17063936);   // 8 KB

    // Prep: f16 conversions + weight swizzles (tiny)
    cvt_f16_kernel<<<2048, 256, 0, stream>>>(z1, z1h, B_ROWS * ML_LEN);
    cvt_f16_kernel<<<2048, 256, 0, stream>>>(z2, z2h, B_ROWS * ML_LEN);
    swizzleB_kernel<<<G_GENE / 32, 128, 0, stream>>>(W0, W0s);
    swizzleB_kernel<<<ML_LEN / 32, 128, 0, stream>>>(W1, W1s);
    swizzleB_kernel<<<ML_LEN / 32, 128, 0, stream>>>(W2, W2s);
    swizzleB_kernel<<<192 / 32, 128, 0, stream>>>(Wa, Was);
    swizzleB_kernel<<<64 / 32, 128, 0, stream>>>(Wb, Wbs);

    // HBM-bound segment-sum over x (the roofline term)
    gene_kernel<<<dim3(B_ROWS / 8, G_GENE / 32), 256, 0, stream>>>(
        x, w_gene, b_gene, geneh);

    // Fused WMMA GEMM chain
    mlp_kernel<<<B_ROWS / 64, 128, 0, stream>>>(
        geneh, z1h, z2h, W0s, W1s, W2s, Was, Wbs,
        b0, b1, b2, ba, bb, Wc, bc, out);
}
